// TwoBodySplineScalarEmbed_30099130811104
// MI455X (gfx1250) — matrix operand order repaired and checked by reading, not compiled, not run on hardware
//
#include <hip/hip_runtime.h>

// ---------------------------------------------------------------------------
// TwoBodySplineScalarEmbed on MI455X (gfx1250, wave32, WMMA)
//
// out[e, 0:64] = sum_b basis[e,b] * coeffs[cls[e], b, :]
// Reformulated as GEMM with class-expanded K = 16 classes * 6 basis = 96:
//   A[16 edges x 96]  (3 nonzero f16 taps per row, staged through LDS)
//   B[96 x 64]        (coeffs as f16, resident in VGPRs in WMMA B layout)
//   D = A @ B via v_wmma_f32_16x16x32_f16 (3 K-steps x 4 N-tiles, f32 accum)
// ---------------------------------------------------------------------------

typedef __attribute__((ext_vector_type(16))) _Float16 v16h;
typedef __attribute__((ext_vector_type(8)))  _Float16 v8h;
typedef __attribute__((ext_vector_type(8)))  float    v8f;

#define NUM_TYPES   4
#define NUM_BASIS   6
#define CHANNELS    64
#define KDIM        96          // NUM_TYPES^2 * NUM_BASIS = 3 * 32
#define ROW_HALVES  104         // 208 bytes/row: 52 dwords -> conflict-free b128
#define WAVES_PER_BLOCK 8

__global__ __launch_bounds__(256) void spline_wmma_kernel(
    const float* __restrict__ x,        // [E]
    const int*   __restrict__ et,       // [2, E]
    const float* __restrict__ coeffs,   // [16, 6, 64] == [96, 64]
    float*       __restrict__ out,      // [E, 64]
    int E)
{
    // Per-wave A staging: 16 rows x 208B (96 f16 cols + pad for LDS banking)
    __shared__ __align__(16) _Float16 stag[WAVES_PER_BLOCK * 16 * ROW_HALVES];

    const int lane  = threadIdx.x & 31;
    const int wInB  = threadIdx.x >> 5;
    const int wid   = blockIdx.x * WAVES_PER_BLOCK + wInB;
    const int nWav  = gridDim.x * WAVES_PER_BLOCK;
    const int m     = lane & 15;       // edge row within tile / N column
    const int hi    = lane >> 4;       // half-wave select

    const int nTiles = (E + 15) >> 4;
    if (wid >= nTiles) return;

    // ---- Preload B (coeffs -> f16) into registers, contiguous-k order (same as A) ----
    // 16-bit B 32x16: lanes 0-15 hold K=0..15, lanes 16-31 hold K=16..31;
    // v16h element i <-> K = kbase + i at column N = 16t + (lane&15).
    v16h B[4][3];
    #pragma unroll
    for (int t = 0; t < 4; ++t) {
        const int n = t * 16 + m;
        #pragma unroll
        for (int s = 0; s < 3; ++s) {
            const int kbase = s * 32 + hi * 16;
            v16h b;
            #pragma unroll
            for (int i = 0; i < 16; ++i)
                b[i] = (_Float16)coeffs[(kbase + i) * CHANNELS + n];
            B[t][s] = b;
        }
    }

    // ---- Zero this wave's staging region (one time) ----
    _Float16* wst = &stag[wInB * 16 * ROW_HALVES];
    {
        unsigned int* p = (unsigned int*)wst;    // 16*104 halves = 832 dwords
        #pragma unroll
        for (int i = 0; i < 26; ++i) p[lane + i * 32] = 0u;
    }

    int prevOff = 0;   // previously written tap offset in my row (lanes 0-15)

    for (int tile = wid; tile < nTiles; tile += nWav) {
        const int e0 = tile * 16;
        const int e  = min(e0 + m, E - 1);       // clamp tail (stores guarded)

        // ---- Per-edge spline: 3 nonzero quadratic B-spline taps ----
        const float xv = x[e];
        const int   cls = et[e] * NUM_TYPES + et[E + e];
        const float xc  = fminf(fmaxf(xv, 0.0f), 1.0f - 1e-6f);
        const float s4  = xc * 4.0f;             // h = 0.25
        int cell = (int)s4; cell = cell > 3 ? 3 : cell;
        const float u  = s4 - (float)cell;
        const float w0 = 0.5f * (1.0f - u) * (1.0f - u);
        const float w2 = 0.5f * u * u;
        const float w1 = 1.0f - w0 - w2;
        const int   off = cls * NUM_BASIS + cell;

        _Float16* row = wst + m * ROW_HALVES;
        if (lane < 16) {
            // clear previous taps, then write the 3 new ones (DS in-order)
            row[prevOff]     = (_Float16)0.0f;
            row[prevOff + 1] = (_Float16)0.0f;
            row[prevOff + 2] = (_Float16)0.0f;
            row[off]     = (_Float16)w0;
            row[off + 1] = (_Float16)w1;
            row[off + 2] = (_Float16)w2;
        }
        prevOff = off;

        // ---- Read A operands, 16x32 f16, contiguous-k order (same as B) ----
        // lanes 0-15: VGPR0-7 = K 0..15
        // lanes 16-31: VGPR0-7 = K 16..31
        v16h A[3];
        #pragma unroll
        for (int s = 0; s < 3; ++s) {
            const int kb = s * 32 + hi * 16;
            const v8h lo = *(const v8h*)(row + kb);        // ds_load_b128
            const v8h hv = *(const v8h*)(row + kb + 8);    // ds_load_b128
            v16h a;
            #pragma unroll
            for (int i = 0; i < 8; ++i) { a[i] = lo[i]; a[i + 8] = hv[i]; }
            A[s] = a;
        }

        // ---- 4 N-tiles x 3 K-steps of v_wmma_f32_16x16x32_f16 ----
        const bool full = (e0 + 16) <= E;
        #pragma unroll
        for (int t = 0; t < 4; ++t) {
            v8f acc = {};
            acc = __builtin_amdgcn_wmma_f32_16x16x32_f16(
                      false, A[0], false, B[t][0], (short)0, acc, false, false);
            acc = __builtin_amdgcn_wmma_f32_16x16x32_f16(
                      false, A[1], false, B[t][1], (short)0, acc, false, false);
            acc = __builtin_amdgcn_wmma_f32_16x16x32_f16(
                      false, A[2], false, B[t][2], (short)0, acc, false, false);

            // D layout: VGPR r -> (M = r + 8*hi, N = 16t + m)
            const int   nCol  = t * 16 + m;
            float*      obase = out + (size_t)(e0 + hi * 8) * CHANNELS + nCol;
            if (full) {
                #pragma unroll
                for (int r = 0; r < 8; ++r)
                    obase[(size_t)r * CHANNELS] = acc[r];
            } else {
                #pragma unroll
                for (int r = 0; r < 8; ++r)
                    if (e0 + hi * 8 + r < E)
                        obase[(size_t)r * CHANNELS] = acc[r];
            }
        }
    }
}

extern "C" void kernel_launch(void* const* d_in, const int* in_sizes, int n_in,
                              void* d_out, int out_size, void* d_ws, size_t ws_size,
                              hipStream_t stream) {
    const float* x      = (const float*)d_in[0];
    const int*   et     = (const int*)d_in[1];
    const float* coeffs = (const float*)d_in[2];
    float*       out    = (float*)d_out;
    const int E = in_sizes[0];

    const int nTiles = (E + 15) / 16;
    int blocks = (nTiles + WAVES_PER_BLOCK - 1) / WAVES_PER_BLOCK;
    if (blocks > 1024) blocks = 1024;   // 8192 persistent waves, grid-stride

    spline_wmma_kernel<<<blocks, 256, 0, stream>>>(x, et, coeffs, out, E);
}
